// MatchingNet_31834297598175
// MI455X (gfx1250) — matrix operand order repaired and checked by reading, not compiled, not run on hardware
//
#include <hip/hip_runtime.h>
#include <math.h>

#define LL   3600
#define DD   256
#define BB   2
#define NP1  3601
#define UVS  3604              /* u/v row stride, padded for float4 alignment */
#define NORM   (-8.8818363f)   /* -ln(7200) */
#define LMU_LAST (-0.6931472f) /* ln(3600)-ln(7200) = -ln2 */
#define SIM_SCALE (1.0f / (256.0f * 256.0f * 0.1f))

typedef __attribute__((ext_vector_type(16))) __bf16 v16bf;
typedef __attribute__((ext_vector_type(8)))  __bf16 v8bf;
typedef __attribute__((ext_vector_type(8)))  float  v8f;

// ---------------- online logsumexp helpers ----------------
static __device__ __forceinline__ void lse_acc(float &m, float &s, float x) {
    if (x > m) { s = s * __expf(m - x) + 1.0f; m = x; }
    else       { s += __expf(x - m); }
}
static __device__ __forceinline__ void lse_merge(float &m, float &s, float m2, float s2) {
    if (s2 <= 0.0f) return;
    if (s  <= 0.0f) { m = m2; s = s2; return; }
    if (m2 > m) { s = s * __expf(m - m2) + s2; m = m2; }
    else        { s += s2 * __expf(m2 - m); }
}

// ---------------- fp32 -> (hi,lo) bf16 split ----------------
__global__ void k_cvt(const float* __restrict__ src, __bf16* __restrict__ hi,
                      __bf16* __restrict__ lo, int n) {
    int i = blockIdx.x * 256 + threadIdx.x;
    if (i >= n) return;
    float x = src[i];
    __bf16 h = (__bf16)x;
    float r  = x - (float)h;
    hi[i] = h;
    lo[i] = (__bf16)r;
}

// ---------------- WMMA GEMM: sim = q @ r^T * scale ----------------
__global__ void __launch_bounds__(256)
k_gemm(const __bf16* __restrict__ qhi, const __bf16* __restrict__ qlo,
       const __bf16* __restrict__ rhi, const __bf16* __restrict__ rlo,
       float* __restrict__ sim) {
    const int wave = threadIdx.x >> 5;
    const int lane = threadIdx.x & 31;
    const int st   = blockIdx.x * 8 + wave;           // s tile
    const int lt   = blockIdx.y;                      // l tile
    const int b    = blockIdx.z;
    if (st >= LL / 16) return;
    const int l0 = lt * 16, s0 = st * 16;
    const int half = lane >> 4;
    const int lrow = lane & 15;
    const size_t qbase = ((size_t)b * LL + (l0 + lrow)) * DD;
    const size_t rbase = ((size_t)b * LL + (s0 + lrow)) * DD;
    const int aoff = half * 8;
    const int boff = half * 8;
    v8f acc = {};
#pragma unroll
    for (int k0 = 0; k0 < DD; k0 += 32) {
        v8bf ah0 = *(const v8bf*)(qhi + qbase + k0 + aoff);
        v8bf ah1 = *(const v8bf*)(qhi + qbase + k0 + 16 + aoff);
        v8bf al0 = *(const v8bf*)(qlo + qbase + k0 + aoff);
        v8bf al1 = *(const v8bf*)(qlo + qbase + k0 + 16 + aoff);
        v16bf ah, al;
#pragma unroll
        for (int e = 0; e < 8; ++e) {
            ah[e] = ah0[e]; ah[e + 8] = ah1[e];
            al[e] = al0[e]; al[e + 8] = al1[e];
        }
        v8bf bh0 = *(const v8bf*)(rhi + rbase + k0 + boff);
        v8bf bh1 = *(const v8bf*)(rhi + rbase + k0 + 16 + boff);
        v8bf bl0 = *(const v8bf*)(rlo + rbase + k0 + boff);
        v8bf bl1 = *(const v8bf*)(rlo + rbase + k0 + 16 + boff);
        v16bf bh, bl;
#pragma unroll
        for (int e = 0; e < 8; ++e) {
            bh[e] = bh0[e]; bh[e + 8] = bh1[e];
            bl[e] = bl0[e]; bl[e + 8] = bl1[e];
        }
        acc = __builtin_amdgcn_wmma_f32_16x16x32_bf16(false, ah, false, bh, (short)0, acc, false, false);
        acc = __builtin_amdgcn_wmma_f32_16x16x32_bf16(false, ah, false, bl, (short)0, acc, false, false);
        acc = __builtin_amdgcn_wmma_f32_16x16x32_bf16(false, al, false, bh, (short)0, acc, false, false);
    }
#pragma unroll
    for (int vv = 0; vv < 8; ++vv) {
        const int M = vv + half * 8;
        sim[((size_t)b * LL + (l0 + M)) * LL + (s0 + lrow)] = acc[vv] * SIM_SCALE;
    }
}

// ---------------- row LSE stats (softmax axis=2), float4 ----------------
__global__ void k_rowlse(const float* __restrict__ sim, float* __restrict__ om,
                         float* __restrict__ os) {
    const int i = blockIdx.x, b = blockIdx.y, tid = threadIdx.x;
    const float4* row4 = (const float4*)(sim + ((size_t)b * LL + i) * LL);
    float m = -INFINITY, s = 0.0f;
    for (int j = tid; j < LL / 4; j += 256) {
        float4 x = row4[j];
        lse_acc(m, s, x.x); lse_acc(m, s, x.y); lse_acc(m, s, x.z); lse_acc(m, s, x.w);
    }
    __shared__ float shm[256], shs[256];
    shm[tid] = m; shs[tid] = s; __syncthreads();
    for (int o = 128; o > 0; o >>= 1) {
        if (tid < o) lse_merge(shm[tid], shs[tid], shm[tid + o], shs[tid + o]);
        __syncthreads();
    }
    if (tid == 0) { om[b * LL + i] = shm[0]; os[b * LL + i] = shs[0]; }
}

// ---------------- col LSE stats: thread owns 4 consecutive cols, b128 reads ----------------
__global__ void k_collse(const float* __restrict__ sim, float* __restrict__ om,
                         float* __restrict__ os) {
    const int tx = threadIdx.x, ty = threadIdx.y;      // (64,4)
    const int j0 = blockIdx.x * 256 + tx * 4;
    const int b  = blockIdx.y;
    float m[4] = {-INFINITY, -INFINITY, -INFINITY, -INFINITY};
    float s[4] = {0.0f, 0.0f, 0.0f, 0.0f};
    if (j0 + 3 < LL) {
        for (int i = ty; i < LL; i += 4) {
            float4 x = *(const float4*)(sim + ((size_t)b * LL + i) * LL + j0);
            lse_acc(m[0], s[0], x.x); lse_acc(m[1], s[1], x.y);
            lse_acc(m[2], s[2], x.z); lse_acc(m[3], s[3], x.w);
        }
    }
    __shared__ float shm[4][256], shs[4][256];
#pragma unroll
    for (int c = 0; c < 4; ++c) { shm[ty][tx * 4 + c] = m[c]; shs[ty][tx * 4 + c] = s[c]; }
    __syncthreads();
    if (ty == 0 && j0 + 3 < LL) {
#pragma unroll
        for (int c = 0; c < 4; ++c) {
            float M = shm[0][tx * 4 + c], S = shs[0][tx * 4 + c];
            for (int t = 1; t < 4; ++t) lse_merge(M, S, shm[t][tx * 4 + c], shs[t][tx * 4 + c]);
            om[b * LL + j0 + c] = M; os[b * LL + j0 + c] = S;
        }
    }
}

// ---------------- cm = softmax(axis=1)*softmax(axis=2), float4 ----------------
__global__ void k_cm(const float* __restrict__ sim,
                     const float* __restrict__ rm, const float* __restrict__ rs,
                     const float* __restrict__ cmx, const float* __restrict__ cs,
                     float* __restrict__ out0) {
    const unsigned NB4 = (unsigned)LL * (LL / 4);           // per-batch float4 count
    const unsigned idx4 = blockIdx.x * 256u + threadIdx.x;
    if (idx4 >= BB * NB4) return;
    const unsigned b = idx4 / NB4, rem = idx4 % NB4;
    const unsigned l = rem / (LL / 4), s4 = rem % (LL / 4);
    float4 x  = ((const float4*)(sim + (size_t)b * LL * LL))[rem];
    float4 cM = ((const float4*)(cmx + b * LL))[s4];
    float4 cS = ((const float4*)(cs  + b * LL))[s4];
    const float rM = rm[b * LL + l], rS = rs[b * LL + l];
    float4 o;
    o.x = __expf((x.x - rM) + (x.x - cM.x)) / (rS * cS.x);
    o.y = __expf((x.y - rM) + (x.y - cM.y)) / (rS * cS.y);
    o.z = __expf((x.z - rM) + (x.z - cM.z)) / (rS * cS.z);
    o.w = __expf((x.w - rM) + (x.w - cM.w)) / (rS * cS.w);
    ((float4*)(out0 + (size_t)b * LL * LL))[rem] = o;
}

// ---------------- plain row/col max of cm, float4 ----------------
__global__ void k_rowmax(const float* __restrict__ a, float* __restrict__ om) {
    const int i = blockIdx.x, b = blockIdx.y, tid = threadIdx.x;
    const float4* row4 = (const float4*)(a + ((size_t)b * LL + i) * LL);
    float m = -INFINITY;
    for (int j = tid; j < LL / 4; j += 256) {
        float4 x = row4[j];
        m = fmaxf(m, fmaxf(fmaxf(x.x, x.y), fmaxf(x.z, x.w)));
    }
    __shared__ float sh[256];
    sh[tid] = m; __syncthreads();
    for (int o = 128; o > 0; o >>= 1) {
        if (tid < o) sh[tid] = fmaxf(sh[tid], sh[tid + o]);
        __syncthreads();
    }
    if (tid == 0) om[b * LL + i] = sh[0];
}
__global__ void k_colmax(const float* __restrict__ a, float* __restrict__ om) {
    const int tx = threadIdx.x, ty = threadIdx.y;      // (64,4)
    const int j0 = blockIdx.x * 256 + tx * 4;
    const int b  = blockIdx.y;
    float m[4] = {-INFINITY, -INFINITY, -INFINITY, -INFINITY};
    if (j0 + 3 < LL) {
        for (int i = ty; i < LL; i += 4) {
            float4 x = *(const float4*)(a + ((size_t)b * LL + i) * LL + j0);
            m[0] = fmaxf(m[0], x.x); m[1] = fmaxf(m[1], x.y);
            m[2] = fmaxf(m[2], x.z); m[3] = fmaxf(m[3], x.w);
        }
    }
    __shared__ float sh[4][256];
#pragma unroll
    for (int c = 0; c < 4; ++c) sh[ty][tx * 4 + c] = m[c];
    __syncthreads();
    if (ty == 0 && j0 + 3 < LL) {
#pragma unroll
        for (int c = 0; c < 4; ++c)
            om[b * LL + j0 + c] = fmaxf(fmaxf(sh[0][tx * 4 + c], sh[1][tx * 4 + c]),
                                        fmaxf(sh[2][tx * 4 + c], sh[3][tx * 4 + c]));
    }
}

// ---------------- cf = cm * mutual-max mask, float4 ----------------
__global__ void k_cf(const float* __restrict__ cm0, const float* __restrict__ rmax,
                     const float* __restrict__ cmax, float* __restrict__ out1) {
    const unsigned NB4 = (unsigned)LL * (LL / 4);
    const unsigned idx4 = blockIdx.x * 256u + threadIdx.x;
    if (idx4 >= BB * NB4) return;
    const unsigned b = idx4 / NB4, rem = idx4 % NB4;
    const unsigned l = rem / (LL / 4), s4 = rem % (LL / 4);
    float4 x  = ((const float4*)(cm0 + (size_t)b * LL * LL))[rem];
    float4 cM = ((const float4*)(cmax + b * LL))[s4];
    const float rM = rmax[b * LL + l];
    float4 o;
    o.x = (x.x == rM && x.x == cM.x) ? x.x : 0.0f;
    o.y = (x.y == rM && x.y == cM.y) ? x.y : 0.0f;
    o.z = (x.z == rM && x.z == cM.z) ? x.z : 0.0f;
    o.w = (x.w == rM && x.w == cM.w) ? x.w : 0.0f;
    ((float4*)(out1 + (size_t)b * LL * LL))[rem] = o;
}

__global__ void k_zero(float* p, int n) {
    int i = blockIdx.x * 256 + threadIdx.x;
    if (i < n) p[i] = 0.0f;
}

// ---------------- Sinkhorn u-update: per-row LSE of Z + v, float4 ----------------
__global__ void k_upd_u(const float* __restrict__ sim, const float* __restrict__ v,
                        float* __restrict__ u, const float* __restrict__ alpha_p) {
    const int i = blockIdx.x, b = blockIdx.y, tid = threadIdx.x;
    const float alpha = alpha_p[0];
    const float* vb = v + b * UVS;
    float m = -INFINITY, s = 0.0f;
    if (i < LL) {
        const float4* row4 = (const float4*)(sim + ((size_t)b * LL + i) * LL);
        const float4* vb4  = (const float4*)vb;
        for (int j = tid; j < LL / 4; j += 256) {
            float4 x = row4[j];
            float4 vv = vb4[j];
            lse_acc(m, s, x.x + vv.x); lse_acc(m, s, x.y + vv.y);
            lse_acc(m, s, x.z + vv.z); lse_acc(m, s, x.w + vv.w);
        }
        if (tid == 0) lse_acc(m, s, alpha + vb[LL]);        // bin column
    } else {                                                 // bin row: all alpha
        for (int j = tid; j < NP1; j += 256) lse_acc(m, s, alpha + vb[j]);
    }
    __shared__ float shm[256], shs[256];
    shm[tid] = m; shs[tid] = s; __syncthreads();
    for (int o = 128; o > 0; o >>= 1) {
        if (tid < o) lse_merge(shm[tid], shs[tid], shm[tid + o], shs[tid + o]);
        __syncthreads();
    }
    if (tid == 0) {
        const float lmu = (i < LL) ? NORM : LMU_LAST;
        u[b * UVS + i] = lmu - (shm[0] + __logf(shs[0]));
    }
}

// ---------------- Sinkhorn v-update: 4 cols/thread, b128 coalesced ----------------
__global__ void k_upd_v(const float* __restrict__ sim, const float* __restrict__ u,
                        float* __restrict__ v, const float* __restrict__ alpha_p) {
    const int tx = threadIdx.x, ty = threadIdx.y;      // (64,4)
    const int j0 = blockIdx.x * 256 + tx * 4;
    const int b  = blockIdx.y;
    const float alpha = alpha_p[0];
    const float* ub = u + b * UVS;
    float m[4] = {-INFINITY, -INFINITY, -INFINITY, -INFINITY};
    float s[4] = {0.0f, 0.0f, 0.0f, 0.0f};
    if (j0 + 3 < LL) {
        for (int i = ty; i < LL; i += 4) {
            float4 x = *(const float4*)(sim + ((size_t)b * LL + i) * LL + j0);
            const float uu = ub[i];
            lse_acc(m[0], s[0], x.x + uu); lse_acc(m[1], s[1], x.y + uu);
            lse_acc(m[2], s[2], x.z + uu); lse_acc(m[3], s[3], x.w + uu);
        }
        if (ty == 0) {                                  // bin row contributes to each col
            const float a = alpha + ub[LL];
            lse_acc(m[0], s[0], a); lse_acc(m[1], s[1], a);
            lse_acc(m[2], s[2], a); lse_acc(m[3], s[3], a);
        }
    } else if (j0 == LL) {                              // bin column: alpha + u over all rows
        for (int i = ty; i < NP1; i += 4) lse_acc(m[0], s[0], alpha + ub[i]);
    }
    __shared__ float shm[4][256], shs[4][256];
#pragma unroll
    for (int c = 0; c < 4; ++c) { shm[ty][tx * 4 + c] = m[c]; shs[ty][tx * 4 + c] = s[c]; }
    __syncthreads();
    if (ty == 0) {
#pragma unroll
        for (int c = 0; c < 4; ++c) {
            const int j = j0 + c;
            if (j > LL) continue;
            if (j == LL && c != 0) continue;            // bin column only in lane c==0 path
            if (j < LL && j0 + 3 >= LL) continue;       // (never: LL%4==0)
            float M = shm[0][tx * 4 + c], S = shs[0][tx * 4 + c];
            for (int t = 1; t < 4; ++t) lse_merge(M, S, shm[t][tx * 4 + c], shs[t][tx * 4 + c]);
            const float lnu = (j < LL) ? NORM : LMU_LAST;
            v[b * UVS + j] = lnu - (M + __logf(S));
        }
    }
}

// ---------------- conf_skh = exp(sim + u + v - norm), in place, float4 ----------------
__global__ void k_final(float* __restrict__ z, const float* __restrict__ u,
                        const float* __restrict__ v) {
    const unsigned NB4 = (unsigned)LL * (LL / 4);
    const unsigned idx4 = blockIdx.x * 256u + threadIdx.x;
    if (idx4 >= BB * NB4) return;
    const unsigned b = idx4 / NB4, rem = idx4 % NB4;
    const unsigned l = rem / (LL / 4), s4 = rem % (LL / 4);
    float4* zp = (float4*)(z + (size_t)b * LL * LL);
    float4 x  = zp[rem];
    float4 vv = ((const float4*)(v + b * UVS))[s4];
    const float uu = u[b * UVS + l] - NORM;
    x.x = __expf(x.x + uu + vv.x);
    x.y = __expf(x.y + uu + vv.y);
    x.z = __expf(x.z + uu + vv.z);
    x.w = __expf(x.w + uu + vv.w);
    zp[rem] = x;
}

extern "C" void kernel_launch(void* const* d_in, const int* in_sizes, int n_in,
                              void* d_out, int out_size, void* d_ws, size_t ws_size,
                              hipStream_t stream) {
    const float* q     = (const float*)d_in[0];
    const float* r     = (const float*)d_in[1];
    const float* alpha = (const float*)d_in[2];

    const size_t NN  = (size_t)BB * LL * LL;
    const size_t NLD = (size_t)BB * LL * DD;
    float* out0 = (float*)d_out;               // cm
    float* out1 = out0 + NN;                   // cf
    float* sim  = out1 + NN;                   // sim -> exp'd in place -> conf_skh

    char* w = (char*)d_ws;
    size_t off = 0;
    auto take = [&](size_t bytes) -> void* {
        void* p = w + off; off += (bytes + 255) & ~(size_t)255; return p;
    };
    __bf16* qhi = (__bf16*)take(NLD * 2);
    __bf16* qlo = (__bf16*)take(NLD * 2);
    __bf16* rhi = (__bf16*)take(NLD * 2);
    __bf16* rlo = (__bf16*)take(NLD * 2);
    float* rowm  = (float*)take((size_t)BB * LL * 4);
    float* rows_ = (float*)take((size_t)BB * LL * 4);
    float* colm  = (float*)take((size_t)BB * LL * 4);
    float* cols_ = (float*)take((size_t)BB * LL * 4);
    float* cmrow = (float*)take((size_t)BB * LL * 4);
    float* cmcol = (float*)take((size_t)BB * LL * 4);
    float* uv = (float*)take((size_t)2 * BB * UVS * 4);
    float* u = uv;
    float* v = uv + (size_t)BB * UVS;

    const int cvtBlocks = (int)((NLD + 255) / 256);
    k_cvt<<<cvtBlocks, 256, 0, stream>>>(q, qhi, qlo, (int)NLD);
    k_cvt<<<cvtBlocks, 256, 0, stream>>>(r, rhi, rlo, (int)NLD);

    k_gemm<<<dim3((LL / 16 + 7) / 8, LL / 16, BB), 256, 0, stream>>>(qhi, qlo, rhi, rlo, sim);

    k_rowlse<<<dim3(LL, BB), 256, 0, stream>>>(sim, rowm, rows_);
    k_collse<<<dim3((LL + 255) / 256, BB), dim3(64, 4), 0, stream>>>(sim, colm, cols_);

    const int ew4Blocks = (int)((NN / 4 + 255) / 256);
    k_cm<<<ew4Blocks, 256, 0, stream>>>(sim, rowm, rows_, colm, cols_, out0);
    k_rowmax<<<dim3(LL, BB), 256, 0, stream>>>(out0, cmrow);
    k_colmax<<<dim3((LL + 255) / 256, BB), dim3(64, 4), 0, stream>>>(out0, cmcol);
    k_cf<<<ew4Blocks, 256, 0, stream>>>(out0, cmrow, cmcol, out1);

    k_zero<<<(2 * BB * UVS + 255) / 256, 256, 0, stream>>>(uv, 2 * BB * UVS);
    for (int it = 0; it < 50; ++it) {
        k_upd_u<<<dim3(NP1, BB), 256, 0, stream>>>(sim, v, u, alpha);
        k_upd_v<<<dim3((NP1 + 255) / 256, BB), dim3(64, 4), 0, stream>>>(sim, u, v, alpha);
    }
    k_final<<<ew4Blocks, 256, 0, stream>>>(sim, u, v);
}
